// FastCriterion_85306640433941
// MI455X (gfx1250) — matrix operand order repaired and checked by reading, not compiled, not run on hardware
//
#include <hip/hip_runtime.h>
#include <hip/hip_bf16.h>
#include <stdint.h>

// ---------------------------------------------------------------------------
// Problem constants (from reference setup_inputs)
// ---------------------------------------------------------------------------
#define BB    8
#define QQ    300
#define CC    300
#define NN    300
#define DD    256
#define WW    96
#define HH    96
#define BQ    (BB*QQ)          // 2400
#define WH    (WW*HH)          // 9216
#define IMG   224.0f
#define IOU_T 0.5f
#define PADK  320              // C/N/Q padded to 320 (=10*32) for WMMA
#define TILE  128              // pixel-tile width in fused dice kernel
#define NT_WH (WH/TILE)        // 72 tiles

typedef __attribute__((ext_vector_type(16))) __bf16 v16bf;
typedef __attribute__((ext_vector_type(8)))  float  v8f;

// ---------------------------------------------------------------------------
// Deterministic hash-based gumbel noise (stand-in for JAX threefry streams)
// ---------------------------------------------------------------------------
__device__ __forceinline__ float gumbelf(uint32_t idx, uint32_t salt) {
  uint32_t x = idx * 0x9E3779B1u + salt * 0x85EBCA77u + 0x165667B1u;
  x ^= x >> 16; x *= 0x7FEB352Du; x ^= x >> 15; x *= 0x846CA68Bu; x ^= x >> 16;
  float u = (float)(x >> 8) * (1.0f / 16777216.0f);
  u = fminf(fmaxf(u, 1e-7f), 1.0f - 1e-7f);
  return -__logf(-__logf(u));
}

// ---------------------------------------------------------------------------
// CDNA5 async global->LDS copy (ASYNCcnt path) -- inline asm per ISA 10/15.18
// ---------------------------------------------------------------------------
__device__ __forceinline__ void async_b128(uint32_t lds_addr, const void* g) {
  asm volatile("global_load_async_to_lds_b128 %0, %1, off"
               :: "v"(lds_addr), "v"(g) : "memory");
}
__device__ __forceinline__ void async_wait0() {
  asm volatile("s_wait_asynccnt 0x0" ::: "memory");
}

// ---------------------------------------------------------------------------
// WMMA fragment loaders (contiguous k order per lane half, wave32)
// A (16x32 bf16): lane half=l>>4, m=l&15; vgpr v holds k=half*16+2v
// ---------------------------------------------------------------------------
__device__ __forceinline__ v16bf load_frag_a(const __bf16* A, int lda,
                                             int row0, int k0, int Mmax) {
  int lane = threadIdx.x & 31;
  int half = lane >> 4;
  int m = row0 + (lane & 15);
  if (m >= Mmax) m = Mmax - 1;
  const __bf16* p = A + (size_t)m * lda + k0;
  v16bf a;
#pragma unroll
  for (int v = 0; v < 8; ++v) {
    int k = half * 16 + 2 * v;
    union { uint32_t u; __bf16 h[2]; } cv;
    cv.u = *(const uint32_t*)(p + k);
    a[2 * v] = cv.h[0]; a[2 * v + 1] = cv.h[1];
  }
  return a;
}

// B (32x16 bf16) from row-major Bt where element(k,n)=Bt[n*ldb+k]
// lane half=l>>4, n=l&15; vgpr v holds k = half*16 + 2v
__device__ __forceinline__ v16bf load_frag_b_nt(const __bf16* Bt, int ldb,
                                                int col0, int k0, int Nmax) {
  int lane = threadIdx.x & 31;
  int half = lane >> 4;
  int n = col0 + (lane & 15);
  if (n >= Nmax) n = Nmax - 1;
  const __bf16* p = Bt + (size_t)n * ldb + k0;
  v16bf b;
#pragma unroll
  for (int v = 0; v < 8; ++v) {
    int k = half * 16 + 2 * v;
    union { uint32_t u; __bf16 h[2]; } cv;
    cv.u = *(const uint32_t*)(p + k);
    b[2 * v] = cv.h[0]; b[2 * v + 1] = cv.h[1];
  }
  return b;
}

// B operand from an LDS tile X laid out [k][TILE] (k-stride = TILE elems)
__device__ __forceinline__ v16bf load_frag_b_lds(const __bf16* X, int col0, int k0) {
  int lane = threadIdx.x & 31;
  int half = lane >> 4;
  int c = col0 + (lane & 15);
  v16bf b;
#pragma unroll
  for (int v = 0; v < 8; ++v) {
    int k = k0 + half * 16 + 2 * v;
    b[2 * v]     = X[(size_t)k * TILE + c];
    b[2 * v + 1] = X[(size_t)(k + 1) * TILE + c];
  }
  return b;
}

__device__ __forceinline__ v8f v8f_zero() {
  v8f z;
#pragma unroll
  for (int i = 0; i < 8; ++i) z[i] = 0.0f;
  return z;
}

// ---------------------------------------------------------------------------
// Generic NT GEMM: C[m][n] = sum_k A[m][k]*Bt[n][k], bf16 in / f32 out.
// One wave per block computes a 16x64 tile (4 accumulators).
// ---------------------------------------------------------------------------
__global__ void gemm_nt_wmma(const __bf16* __restrict__ A,
                             const __bf16* __restrict__ Bt,
                             float* __restrict__ Cmat,
                             int M, int N, int K, int lda, int ldb, int ldc) {
  int row0 = blockIdx.x * 16;
  int col0 = blockIdx.y * 64;
  v8f acc0 = v8f_zero(), acc1 = v8f_zero(), acc2 = v8f_zero(), acc3 = v8f_zero();
  for (int k0 = 0; k0 < K; k0 += 32) {
    __builtin_prefetch(A + (size_t)row0 * lda + k0 + 64, 0, 1);   // global_prefetch
    v16bf af = load_frag_a(A, lda, row0, k0, M);
    v16bf b0 = load_frag_b_nt(Bt, ldb, col0 +  0, k0, N);
    acc0 = __builtin_amdgcn_wmma_f32_16x16x32_bf16(false, af, false, b0, (short)0, acc0, false, false);
    v16bf b1 = load_frag_b_nt(Bt, ldb, col0 + 16, k0, N);
    acc1 = __builtin_amdgcn_wmma_f32_16x16x32_bf16(false, af, false, b1, (short)0, acc1, false, false);
    v16bf b2 = load_frag_b_nt(Bt, ldb, col0 + 32, k0, N);
    acc2 = __builtin_amdgcn_wmma_f32_16x16x32_bf16(false, af, false, b2, (short)0, acc2, false, false);
    v16bf b3 = load_frag_b_nt(Bt, ldb, col0 + 48, k0, N);
    acc3 = __builtin_amdgcn_wmma_f32_16x16x32_bf16(false, af, false, b3, (short)0, acc3, false, false);
  }
  int lane = threadIdx.x & 31, half = lane >> 4, nn = lane & 15;
  v8f accs[4] = {acc0, acc1, acc2, acc3};
#pragma unroll
  for (int c = 0; c < 4; ++c)
#pragma unroll
    for (int r = 0; r < 8; ++r) {
      int m = row0 + r + 8 * half, n = col0 + c * 16 + nn;
      if (m < M && n < N) Cmat[(size_t)m * ldc + n] = accs[c][r];
    }
}

// ---------------------------------------------------------------------------
// f32 -> bf16 conversion
// ---------------------------------------------------------------------------
__global__ void convert_bf16(const float* __restrict__ src, __bf16* __restrict__ dst, int n) {
  int i = blockIdx.x * blockDim.x + threadIdx.x;
  if (i < n) dst[i] = (__bf16)src[i];
}

// ---------------------------------------------------------------------------
// out_scores = max_c sim; ox = IMG*argmax_c(sim+g); out_bbox = xyxy + offsets
// ---------------------------------------------------------------------------
__global__ void boxes_kernel(const float* __restrict__ sim,
                             const float* __restrict__ pred_boxes,
                             float* __restrict__ out_bbox,
                             float* __restrict__ out_scores) {
  int m = blockIdx.x * blockDim.x + threadIdx.x;
  if (m >= BQ) return;
  float mx = -1e30f, best = -1e30f;
  int bi = 0;
  for (int c = 0; c < CC; ++c) {
    float v = sim[(size_t)m * CC + c];
    mx = fmaxf(mx, v);
    float g = v + gumbelf((uint32_t)(m * CC + c), 0xA001u);
    if (g > best) { best = g; bi = c; }
  }
  out_scores[m] = mx;
  float ox = IMG * (float)bi;
  float oy = IMG * (float)(m / QQ);
  const float* pb = pred_boxes + (size_t)m * 4;
  float cx = pb[0], cy = pb[1], w = pb[2], h = pb[3];
  out_bbox[m * 4 + 0] = cx - 0.5f * w + ox;
  out_bbox[m * 4 + 1] = cy - 0.5f * h + oy;
  out_bbox[m * 4 + 2] = cx + 0.5f * w + ox;
  out_bbox[m * 4 + 3] = cy + 0.5f * h + oy;
}

__global__ void tb_kernel(const float* __restrict__ tgt_bbox,
                          const int* __restrict__ tgt_ids,
                          const int* __restrict__ batch_idx,
                          float* __restrict__ tb) {
  int n = blockIdx.x * blockDim.x + threadIdx.x;
  if (n >= NN) return;
  const float* b = tgt_bbox + (size_t)n * 4;
  float cx = b[0], cy = b[1], w = b[2], h = b[3];
  float ox = IMG * (float)tgt_ids[n], oy = IMG * (float)batch_idx[n];
  tb[n * 4 + 0] = cx - 0.5f * w + ox;
  tb[n * 4 + 1] = cy - 0.5f * h + oy;
  tb[n * 4 + 2] = cx + 0.5f * w + ox;
  tb[n * 4 + 3] = cy + 0.5f * h + oy;
}

// ---------------------------------------------------------------------------
// Greedy NMS, single block of 1024 threads (O(n^2) rank sort + serial sweep)
// ---------------------------------------------------------------------------
__global__ void nms_kernel(const float* __restrict__ boxes,
                           const float* __restrict__ scores,
                           int* __restrict__ keepIdx, int* __restrict__ Kout) {
  __shared__ float bx0[BQ], by0[BQ], bx1[BQ], by1[BQ], ar[BQ];
  __shared__ int   ord[BQ];
  __shared__ char  sup[BQ];
  int t = threadIdx.x, nt = blockDim.x;
  for (int i = t; i < BQ; i += nt) {
    float si = scores[i];
    int cnt = 0;
    for (int j = 0; j < BQ; ++j) {
      float sj = scores[j];
      cnt += (sj > si) || (sj == si && j < i);
    }
    ord[cnt] = i;
  }
  __syncthreads();
  for (int r = t; r < BQ; r += nt) {
    int idx = ord[r];
    float x0 = boxes[idx * 4], y0 = boxes[idx * 4 + 1];
    float x1 = boxes[idx * 4 + 2], y1 = boxes[idx * 4 + 3];
    bx0[r] = x0; by0[r] = y0; bx1[r] = x1; by1[r] = y1;
    ar[r] = (x1 - x0) * (y1 - y0);
    sup[r] = 0;
  }
  __syncthreads();
  for (int i = 0; i < BQ; ++i) {
    __syncthreads();
    if (!sup[i]) {
      float x0 = bx0[i], y0 = by0[i], x1 = bx1[i], y1 = by1[i], ai = ar[i];
      for (int j = i + 1 + t; j < BQ; j += nt) {
        float lx = fmaxf(x0, bx0[j]), ly = fmaxf(y0, by0[j]);
        float rx = fminf(x1, bx1[j]), ry = fminf(y1, by1[j]);
        float iw = fmaxf(rx - lx, 0.0f), ih = fmaxf(ry - ly, 0.0f);
        float inter = iw * ih;
        float iou = inter / (ai + ar[j] - inter);
        if (iou > IOU_T) sup[j] = 1;
      }
    }
  }
  __syncthreads();
  if (t == 0) {
    int K = 0;
    for (int i = 0; i < BQ; ++i) if (!sup[i]) keepIdx[K++] = ord[i];
    *Kout = K;
  }
}

// ---------------------------------------------------------------------------
// Per-kept-box: IoU row vs tb, softmax, hard-gumbel argmax -> scatter add
// ---------------------------------------------------------------------------
__global__ void iou_scatter(const float* __restrict__ out_bbox,
                            const float* __restrict__ tb,
                            const int* __restrict__ keepIdx,
                            const int* __restrict__ Kp,
                            const float* __restrict__ plog,
                            float* __restrict__ gt_sel,
                            float* __restrict__ out_embs) {
  int k = blockIdx.x;
  if (k >= *Kp) return;
  int src = keepIdx[k];
  float kx0 = out_bbox[src * 4], ky0 = out_bbox[src * 4 + 1];
  float kx1 = out_bbox[src * 4 + 2], ky1 = out_bbox[src * 4 + 3];
  float ak = (kx1 - kx0) * (ky1 - ky0);
  __shared__ float p[NN];
  __shared__ float redf[512];
  __shared__ int   redi[512];
  __shared__ int   nstar;
  int t = threadIdx.x;
  if (t < NN) {
    float x0 = tb[t * 4], y0 = tb[t * 4 + 1], x1 = tb[t * 4 + 2], y1 = tb[t * 4 + 3];
    float an = (x1 - x0) * (y1 - y0);
    float lx = fmaxf(kx0, x0), ly = fmaxf(ky0, y0);
    float rx = fminf(kx1, x1), ry = fminf(ky1, y1);
    float inter = fmaxf(rx - lx, 0.0f) * fmaxf(ry - ly, 0.0f);
    float iou = inter / (ak + an - inter);
    if (isnan(iou)) iou = 0.0f;
    p[t] = iou;
  }
  __syncthreads();
  // softmax over 300
  float mv = (t < NN) ? p[t] : -1e30f;
  redf[t] = mv; __syncthreads();
  for (int s = 256; s > 0; s >>= 1) { if (t < s) redf[t] = fmaxf(redf[t], redf[t + s]); __syncthreads(); }
  float mx = redf[0]; __syncthreads();
  float ev = (t < NN) ? __expf(p[t] - mx) : 0.0f;
  redf[t] = ev; __syncthreads();
  for (int s = 256; s > 0; s >>= 1) { if (t < s) redf[t] += redf[t + s]; __syncthreads(); }
  float sum = redf[0]; __syncthreads();
  float val = (t < NN) ? (ev / sum + gumbelf((uint32_t)(k * NN + t), 0xB002u)) : -1e30f;
  redf[t] = val; redi[t] = (t < NN) ? t : 0; __syncthreads();
  for (int s = 256; s > 0; s >>= 1) {
    if (t < s && redf[t + s] > redf[t]) { redf[t] = redf[t + s]; redi[t] = redi[t + s]; }
    __syncthreads();
  }
  if (t == 0) nstar = redi[0];
  __syncthreads();
  int n = nstar;
  if (t < DD) atomicAdd(&out_embs[(size_t)n * DD + t], plog[(size_t)src * DD + t]);
  if (t < 4)  atomicAdd(&gt_sel[n * 4 + t], out_bbox[src * 4 + t]);
}

// ---------------------------------------------------------------------------
// CE-diag of M and M^T: per-i row/col logsumexp, accumulate diag log-softmax
// ---------------------------------------------------------------------------
__global__ void ce_kernel(const float* __restrict__ Mm, float* __restrict__ accCe) {
  int i = blockIdx.x, t = threadIdx.x;
  __shared__ float red[256];
  float mr = -1e30f, mc = -1e30f;
  for (int j = t; j < NN; j += 256) {
    mr = fmaxf(mr, Mm[(size_t)i * NN + j]);
    mc = fmaxf(mc, Mm[(size_t)j * NN + i]);
  }
  red[t] = mr; __syncthreads();
  for (int s = 128; s > 0; s >>= 1) { if (t < s) red[t] = fmaxf(red[t], red[t + s]); __syncthreads(); }
  float rowmax = red[0]; __syncthreads();
  red[t] = mc; __syncthreads();
  for (int s = 128; s > 0; s >>= 1) { if (t < s) red[t] = fmaxf(red[t], red[t + s]); __syncthreads(); }
  float colmax = red[0]; __syncthreads();
  float sr = 0.0f, sc = 0.0f;
  for (int j = t; j < NN; j += 256) {
    sr += __expf(Mm[(size_t)i * NN + j] - rowmax);
    sc += __expf(Mm[(size_t)j * NN + i] - colmax);
  }
  red[t] = sr; __syncthreads();
  for (int s = 128; s > 0; s >>= 1) { if (t < s) red[t] += red[t + s]; __syncthreads(); }
  float rowsum = red[0]; __syncthreads();
  red[t] = sc; __syncthreads();
  for (int s = 128; s > 0; s >>= 1) { if (t < s) red[t] += red[t + s]; __syncthreads(); }
  float colsum = red[0]; __syncthreads();
  if (t == 0) {
    float d = Mm[(size_t)i * NN + i];
    atomicAdd(accCe, (d - rowmax - __logf(rowsum)) + (d - colmax - __logf(colsum)));
  }
}

// ---------------------------------------------------------------------------
// Column-wise hard gumbel argmax over scores2[:,n] (axis=0) -> sel[n]
// ---------------------------------------------------------------------------
__global__ void colargmax_kernel(const float* __restrict__ sc2, int* __restrict__ sel) {
  int n = blockIdx.x, t = threadIdx.x;
  __shared__ float redf[256];
  __shared__ int   redi[256];
  float best = -1e30f; int bi = 0;
  for (int m = t; m < BQ; m += 256) {
    float v = sc2[(size_t)m * NN + n] + gumbelf((uint32_t)(m * NN + n), 0xC003u);
    if (v > best) { best = v; bi = m; }
  }
  redf[t] = best; redi[t] = bi; __syncthreads();
  for (int s = 128; s > 0; s >>= 1) {
    if (t < s && redf[t + s] > redf[t]) { redf[t] = redf[t + s]; redi[t] = redi[t + s]; }
    __syncthreads();
  }
  if (t == 0) sel[n] = redi[0];
}

// ---------------------------------------------------------------------------
// Gather om rows (= hard one-hot @ pred_masks), emit bf16 om/tgt tiles, MSE
// ---------------------------------------------------------------------------
__global__ void gather_mse(const int* __restrict__ sel,
                           const float* __restrict__ pred_masks,
                           const float* __restrict__ tgt_masks,
                           __bf16* __restrict__ om_bf,
                           __bf16* __restrict__ tm_bf,
                           float* __restrict__ accMse) {
  int n = blockIdx.y;
  int px = blockIdx.x * 256 + threadIdx.x;
  int s = sel[n];
  float p = pred_masks[(size_t)s * WH + px];
  float tm = tgt_masks[(size_t)n * WH + px];
  om_bf[(size_t)n * WH + px] = (__bf16)p;
  tm_bf[(size_t)n * WH + px] = (__bf16)tm;
  float d = p - tm;
  __shared__ float red[256];
  int t = threadIdx.x;
  red[t] = d * d; __syncthreads();
  for (int ss = 128; ss > 0; ss >>= 1) { if (t < ss) red[t] += red[t + ss]; __syncthreads(); }
  if (t == 0) atomicAdd(accMse, red[0]);
}

// ---------------------------------------------------------------------------
// Build padded bf16 A-operands: simA[b][c][q] = sim[b,q,c]/W ; mgs[b][c][n]
// ---------------------------------------------------------------------------
__global__ void build_simA(const float* __restrict__ sim, __bf16* __restrict__ simA) {
  size_t i = (size_t)blockIdx.x * 256 + threadIdx.x;
  if (i >= (size_t)BB * PADK * PADK) return;
  int b = (int)(i / (PADK * PADK));
  int r = (int)(i % (PADK * PADK));
  int c = r / PADK, q = r % PADK;
  float v = 0.0f;
  if (c < CC && q < QQ) v = sim[((size_t)b * QQ + q) * CC + c] * (1.0f / (float)WW);
  simA[i] = (__bf16)v;
}

__global__ void build_mgs(const float* __restrict__ gt_sim,
                          const int* __restrict__ batch_idx,
                          __bf16* __restrict__ mgs) {
  size_t i = (size_t)blockIdx.x * 256 + threadIdx.x;
  if (i >= (size_t)BB * PADK * PADK) return;
  int b = (int)(i / (PADK * PADK));
  int r = (int)(i % (PADK * PADK));
  int c = r / PADK, n = r % PADK;
  float v = 0.0f;
  if (c < CC && n < NN && batch_idx[n] == b) v = gt_sim[(size_t)n * CC + c];
  mgs[i] = (__bf16)v;
}

// ---------------------------------------------------------------------------
// Fused dice kernel: per (pixel tile, batch) compute ocm & gt_cm columns with
// WMMA, softmax over C in LDS, accumulate dice sums. LDS: O(80K) X(80K) Y(80K)
// B-operand tiles staged via global_load_async_to_lds_b128 (ASYNCcnt).
// ---------------------------------------------------------------------------
#define OFF_O 0u
#define OFF_X 81920u
#define OFF_Y 163840u

__device__ __forceinline__ void stage_tile_async(uint32_t lds_region,
                                                 const __bf16* gbase, int col0) {
  int t = threadIdx.x;              // 256 threads
  int lane16 = t & 15;              // 16B chunk within a 256B row
  int rowInPass = t >> 4;           // 16 rows per pass
  uint32_t lds_base = (uint32_t)__builtin_amdgcn_groupstaticsize() + lds_region;
#pragma unroll
  for (int pass = 0; pass < 20; ++pass) {
    int row = pass * 16 + rowInPass;  // 0..319
    const void* g = (const void*)((const char*)gbase +
                     ((size_t)row * WH + col0) * 2 + (size_t)lane16 * 16);
    async_b128(lds_base + (uint32_t)row * 256u + (uint32_t)lane16 * 16u, g);
  }
  async_wait0();
}

__global__ void dice_kernel(const __bf16* __restrict__ simA,
                            const __bf16* __restrict__ mgs,
                            const __bf16* __restrict__ om_bf,
                            const __bf16* __restrict__ tm_bf,
                            float* __restrict__ acc) {
  extern __shared__ __align__(16) char smem[];
  __bf16* Olds = (__bf16*)(smem + OFF_O);
  __bf16* Xlds = (__bf16*)(smem + OFF_X);
  __bf16* Ylds = (__bf16*)(smem + OFF_Y);
  __shared__ float red[256];

  int tile = blockIdx.x;            // 0..71
  int b    = blockIdx.y;            // 0..7
  int col0 = tile * TILE;
  int wave = threadIdx.x >> 5;      // 0..7
  int lane = threadIdx.x & 31, half = lane >> 4, nn = lane & 15;

  // ---- phase 1: O = (sim_b/W) @ om_tile --------------------------------
  stage_tile_async(OFF_X, om_bf, col0);
  __syncthreads();
  const __bf16* A1 = simA + (size_t)b * PADK * PADK;
  for (int s = wave; s < PADK / 16; s += 8) {
    v8f acc8[8];
#pragma unroll
    for (int ct = 0; ct < 8; ++ct) acc8[ct] = v8f_zero();
    for (int kt = 0; kt < PADK / 32; ++kt) {
      v16bf af = load_frag_a(A1, PADK, s * 16, kt * 32, PADK);
#pragma unroll
      for (int ct = 0; ct < 8; ++ct) {
        v16bf bf = load_frag_b_lds(Xlds, ct * 16, kt * 32);
        acc8[ct] = __builtin_amdgcn_wmma_f32_16x16x32_bf16(
            false, af, false, bf, (short)0, acc8[ct], false, false);
      }
    }
#pragma unroll
    for (int ct = 0; ct < 8; ++ct)
#pragma unroll
      for (int r = 0; r < 8; ++r) {
        int row = s * 16 + r + 8 * half;
        Olds[(size_t)row * TILE + ct * 16 + nn] = (__bf16)acc8[ct][r];
      }
  }
  __syncthreads();

  // ---- phase 2: G = mgs_b @ tgt_masks_tile ----------------------------
  stage_tile_async(OFF_X, tm_bf, col0);
  __syncthreads();
  const __bf16* A2 = mgs + (size_t)b * PADK * PADK;
  for (int s = wave; s < PADK / 16; s += 8) {
    v8f acc8[8];
#pragma unroll
    for (int ct = 0; ct < 8; ++ct) acc8[ct] = v8f_zero();
    for (int kt = 0; kt < PADK / 32; ++kt) {
      v16bf af = load_frag_a(A2, PADK, s * 16, kt * 32, PADK);
#pragma unroll
      for (int ct = 0; ct < 8; ++ct) {
        v16bf bf = load_frag_b_lds(Xlds, ct * 16, kt * 32);
        acc8[ct] = __builtin_amdgcn_wmma_f32_16x16x32_bf16(
            false, af, false, bf, (short)0, acc8[ct], false, false);
      }
    }
#pragma unroll
    for (int ct = 0; ct < 8; ++ct)
#pragma unroll
      for (int r = 0; r < 8; ++r) {
        int row = s * 16 + r + 8 * half;
        Ylds[(size_t)row * TILE + ct * 16 + nn] = (__bf16)acc8[ct][r];
      }
  }
  __syncthreads();

  // ---- phase 3: softmax over C per column + dice accumulation ----------
  int t = threadIdx.x;
  float Sp = 0.0f, sI = 0.0f, sT = 0.0f;
  if (t < TILE) {
    float mx = -1e30f;
    for (int c = 0; c < CC; ++c) mx = fmaxf(mx, (float)Olds[(size_t)c * TILE + t]);
    float se = 0.0f;
    for (int c = 0; c < CC; ++c) se += __expf((float)Olds[(size_t)c * TILE + t] - mx);
    float inv = 1.0f / se;
    for (int c = 0; c < CC; ++c) {
      float p = __expf((float)Olds[(size_t)c * TILE + t] - mx) * inv;
      float g = (float)Ylds[(size_t)c * TILE + t];
      Sp += p * g; sI += p; sT += g;
    }
  }
  red[t] = Sp; __syncthreads();
  for (int s = 128; s > 0; s >>= 1) { if (t < s) red[t] += red[t + s]; __syncthreads(); }
  if (t == 0) atomicAdd(&acc[0], red[0]);
  __syncthreads();
  red[t] = sI; __syncthreads();
  for (int s = 128; s > 0; s >>= 1) { if (t < s) red[t] += red[t + s]; __syncthreads(); }
  if (t == 0) atomicAdd(&acc[1 + b], red[0]);
  __syncthreads();
  red[t] = sT; __syncthreads();
  for (int s = 128; s > 0; s >>= 1) { if (t < s) red[t] += red[t + s]; __syncthreads(); }
  if (t == 0) atomicAdd(&acc[9 + b], red[0]);
}

// ---------------------------------------------------------------------------
// gt_ious + L1 over gt_sel vs tb
// ---------------------------------------------------------------------------
__global__ void gtiou_kernel(const float* __restrict__ gt_sel,
                             const float* __restrict__ tb,
                             float* __restrict__ acc) {
  __shared__ float red[512];
  int t = threadIdx.x;
  float iou = 0.0f, l1 = 0.0f;
  if (t < NN) {
    float ax0 = gt_sel[t * 4], ay0 = gt_sel[t * 4 + 1];
    float ax1 = gt_sel[t * 4 + 2], ay1 = gt_sel[t * 4 + 3];
    float bx0 = tb[t * 4], by0 = tb[t * 4 + 1];
    float bx1 = tb[t * 4 + 2], by1 = tb[t * 4 + 3];
    float aa = (ax1 - ax0) * (ay1 - ay0);
    float ab = (bx1 - bx0) * (by1 - by0);
    float inter = fmaxf(fminf(ax1, bx1) - fmaxf(ax0, bx0), 0.0f) *
                  fmaxf(fminf(ay1, by1) - fmaxf(ay0, by0), 0.0f);
    iou = inter / (aa + ab - inter);
    l1 = fabsf(ax0 - bx0) + fabsf(ay0 - by0) + fabsf(ax1 - bx1) + fabsf(ay1 - by1);
  }
  red[t] = iou; __syncthreads();
  for (int s = 256; s > 0; s >>= 1) { if (t < s) red[t] += red[t + s]; __syncthreads(); }
  if (t == 0) atomicAdd(&acc[19], red[0]);
  __syncthreads();
  red[t] = l1; __syncthreads();
  for (int s = 256; s > 0; s >>= 1) { if (t < s) red[t] += red[t + s]; __syncthreads(); }
  if (t == 0) atomicAdd(&acc[20], red[0]);
}

// ---------------------------------------------------------------------------
// Finalize the 5 outputs
// ---------------------------------------------------------------------------
__global__ void finalize_kernel(const float* __restrict__ acc,
                                const int* __restrict__ Kp,
                                float* __restrict__ out) {
  float Kf = (float)(*Kp);
  out[0] = -acc[18] / (float)NN;                                 // class_loss
  float dice = 0.0f;
  for (int b = 0; b < BB; ++b)
    dice += acc[0] / (acc[1 + b] + acc[9 + b] + 1e-6f);
  out[1] = dice / (float)BB;                                      // dice
  out[2] = acc[17] / ((float)NN * (float)WH) / Kf;                // mask loss / K
  out[3] = ((float)NN - acc[19]) / Kf;                            // (1-iou).sum/K
  out[4] = acc[20] / Kf;                                          // l1 / K
}

// ---------------------------------------------------------------------------
// Workspace layout
// ---------------------------------------------------------------------------
static constexpr size_t alignup(size_t x) { return (x + 255) & ~size_t(255); }
static constexpr size_t OFF_ABF   = 0;
static constexpr size_t OFF_CEBF  = OFF_ABF   + alignup((size_t)BQ * DD * 2);
static constexpr size_t OFF_TEBF  = OFF_CEBF  + alignup((size_t)CC * DD * 2);
static constexpr size_t OFF_OEBF  = OFF_TEBF  + alignup((size_t)NN * DD * 2);
static constexpr size_t OFF_SIM   = OFF_OEBF  + alignup((size_t)NN * DD * 2);
static constexpr size_t OFF_SC2   = OFF_SIM   + alignup((size_t)BQ * CC * 4);
static constexpr size_t OFF_GTSIM = OFF_SC2   + alignup((size_t)BQ * NN * 4);
static constexpr size_t OFF_MMAT  = OFF_GTSIM + alignup((size_t)NN * CC * 4);
static constexpr size_t OFF_SIMA  = OFF_MMAT  + alignup((size_t)NN * NN * 4);
static constexpr size_t OFF_MGS   = OFF_SIMA  + alignup((size_t)BB * PADK * PADK * 2);
static constexpr size_t OFF_OMBF  = OFF_MGS   + alignup((size_t)BB * PADK * PADK * 2);
static constexpr size_t OFF_TMBF  = OFF_OMBF  + alignup((size_t)PADK * WH * 2);
static constexpr size_t OFF_OUTBB = OFF_TMBF  + alignup((size_t)PADK * WH * 2);
static constexpr size_t OFF_OSC   = OFF_OUTBB + alignup((size_t)BQ * 4 * 4);
static constexpr size_t OFF_TB    = OFF_OSC   + alignup((size_t)BQ * 4);
static constexpr size_t OFF_GTSEL = OFF_TB    + alignup((size_t)NN * 4 * 4);   // zero region start
static constexpr size_t OFF_OUTE  = OFF_GTSEL + alignup((size_t)NN * 4 * 4);
static constexpr size_t OFF_ACC   = OFF_OUTE  + alignup((size_t)NN * DD * 4);
static constexpr size_t OFF_KEEP  = OFF_ACC   + 256;                            // zero region end
static constexpr size_t OFF_KVAL  = OFF_KEEP  + alignup((size_t)BQ * 4);
static constexpr size_t OFF_SEL   = OFF_KVAL  + 256;

extern "C" void kernel_launch(void* const* d_in, const int* in_sizes, int n_in,
                              void* d_out, int out_size, void* d_ws, size_t ws_size,
                              hipStream_t stream) {
  (void)in_sizes; (void)n_in; (void)out_size; (void)ws_size;
  const float* classenc   = (const float*)d_in[0];
  const float* pred_logits= (const float*)d_in[1];
  const float* pred_boxes = (const float*)d_in[2];
  const float* pred_masks = (const float*)d_in[3];
  const float* tgt_masks  = (const float*)d_in[4];
  const float* tgt_embs   = (const float*)d_in[5];
  const float* tgt_bbox   = (const float*)d_in[6];
  const int*   tgt_ids    = (const int*)d_in[8];
  const int*   batch_idx  = (const int*)d_in[9];
  float* out = (float*)d_out;
  char* ws = (char*)d_ws;

  __bf16* A_bf   = (__bf16*)(ws + OFF_ABF);
  __bf16* CE_bf  = (__bf16*)(ws + OFF_CEBF);
  __bf16* TE_bf  = (__bf16*)(ws + OFF_TEBF);
  __bf16* OE_bf  = (__bf16*)(ws + OFF_OEBF);
  float*  sim    = (float*)(ws + OFF_SIM);
  float*  sc2    = (float*)(ws + OFF_SC2);
  float*  gt_sim = (float*)(ws + OFF_GTSIM);
  float*  Mmat   = (float*)(ws + OFF_MMAT);
  __bf16* simA   = (__bf16*)(ws + OFF_SIMA);
  __bf16* mgs    = (__bf16*)(ws + OFF_MGS);
  __bf16* om_bf  = (__bf16*)(ws + OFF_OMBF);
  __bf16* tm_bf  = (__bf16*)(ws + OFF_TMBF);
  float*  outbb  = (float*)(ws + OFF_OUTBB);
  float*  osc    = (float*)(ws + OFF_OSC);
  float*  tb     = (float*)(ws + OFF_TB);
  float*  gt_sel = (float*)(ws + OFF_GTSEL);
  float*  out_embs = (float*)(ws + OFF_OUTE);
  float*  acc    = (float*)(ws + OFF_ACC);
  int*    keep   = (int*)(ws + OFF_KEEP);
  int*    Kval   = (int*)(ws + OFF_KVAL);
  int*    sel    = (int*)(ws + OFF_SEL);

  // 0) zero accumulators / scatter targets / padded bf16 mask buffers
  hipMemsetAsync(ws + OFF_GTSEL, 0, (OFF_KEEP - OFF_GTSEL), stream);
  hipMemsetAsync(ws + OFF_OMBF, 0, (size_t)PADK * WH * 2, stream);
  hipMemsetAsync(ws + OFF_TMBF, 0, (size_t)PADK * WH * 2, stream);

  // 1) f32 -> bf16 conversions
  convert_bf16<<<(BQ * DD + 255) / 256, 256, 0, stream>>>(pred_logits, A_bf, BQ * DD);
  convert_bf16<<<(CC * DD + 255) / 256, 256, 0, stream>>>(classenc, CE_bf, CC * DD);
  convert_bf16<<<(NN * DD + 255) / 256, 256, 0, stream>>>(tgt_embs, TE_bf, NN * DD);

  // 2) embedding GEMMs (WMMA bf16)
  gemm_nt_wmma<<<dim3(BQ / 16, 5), 32, 0, stream>>>(A_bf, CE_bf, sim, BQ, CC, DD, DD, DD, CC);
  gemm_nt_wmma<<<dim3(BQ / 16, 5), 32, 0, stream>>>(A_bf, TE_bf, sc2, BQ, NN, DD, DD, DD, NN);
  gemm_nt_wmma<<<dim3(19, 5), 32, 0, stream>>>(TE_bf, CE_bf, gt_sim, NN, CC, DD, DD, DD, CC);

  // 3) boxes / targets / NMS
  boxes_kernel<<<(BQ + 255) / 256, 256, 0, stream>>>(sim, pred_boxes, outbb, osc);
  tb_kernel<<<2, 256, 0, stream>>>(tgt_bbox, tgt_ids, batch_idx, tb);
  nms_kernel<<<1, 1024, 0, stream>>>(outbb, osc, keep, Kval);

  // 4) per-kept IoU + hard gumbel assignment -> scatter gt_sel / out_embs
  iou_scatter<<<BQ, 512, 0, stream>>>(outbb, tb, keep, Kval, pred_logits, gt_sel, out_embs);

  // 5) class loss: M = tgt_embs @ out_embs^T, ce both ways
  convert_bf16<<<(NN * DD + 255) / 256, 256, 0, stream>>>(out_embs, OE_bf, NN * DD);
  gemm_nt_wmma<<<dim3(19, 5), 32, 0, stream>>>(TE_bf, OE_bf, Mmat, NN, NN, DD, DD, DD, NN);
  ce_kernel<<<NN, 256, 0, stream>>>(Mmat, &acc[18]);

  // 6) mask assignment (axis-0 hard gumbel) + gather + MSE
  colargmax_kernel<<<NN, 256, 0, stream>>>(sc2, sel);
  gather_mse<<<dim3(WH / 256, NN), 256, 0, stream>>>(sel, pred_masks, tgt_masks,
                                                     om_bf, tm_bf, &acc[17]);

  // 7) fused ocm/gt_cm + softmax(C) + dice (WMMA + async LDS staging)
  build_simA<<<(BB * PADK * PADK + 255) / 256, 256, 0, stream>>>(sim, simA);
  build_mgs<<<(BB * PADK * PADK + 255) / 256, 256, 0, stream>>>(gt_sim, batch_idx, mgs);
  dice_kernel<<<dim3(NT_WH, BB), 256, 245760, 0 ? nullptr : stream>>>(simA, mgs, om_bf, tm_bf, acc);

  // 8) box regression terms + final outputs
  gtiou_kernel<<<1, 512, 0, stream>>>(gt_sel, tb, acc);
  finalize_kernel<<<1, 1, 0, stream>>>(acc, Kval, out);
}
